// Performer_74199855005716
// MI455X (gfx1250) — matrix operand order repaired and checked by reading, not compiled, not run on hardware
//
#include <hip/hip_runtime.h>

// ---------------- problem constants ----------------
#define B_     8
#define N_     4096
#define D_     3
#define L_     6
#define H_     3
#define DH_    128
#define INNER_ 384
#define M_     620
#define MP_    640          // M padded to multiple of 32 (zero features)
#define FF_    12
#define BH_    (B_ * H_)

#define DN_   0.29730177875068026f   // 128^-0.25
#define KM_   0.04016096644512494f   // 620^-0.5
#define EPS_  1e-4f

typedef __attribute__((ext_vector_type(16))) __bf16 v16bf;
typedef __attribute__((ext_vector_type(8)))  float  v8f;
typedef __attribute__((ext_vector_type(4)))  unsigned int u32x4;
typedef __attribute__((ext_vector_type(4)))  int  i32x4;
typedef __attribute__((ext_vector_type(8)))  int  i32x8;

struct Frag32B { uint4 lo, hi; };
union  U8H     { uint4 u; __bf16 h[8]; };

static __device__ __forceinline__ v8f zero8() {
    v8f z = {0.f, 0.f, 0.f, 0.f, 0.f, 0.f, 0.f, 0.f};
    return z;
}

// A-matrix fragment (16x32 bf16): lane holds row (lane&15).
// lane<16: K = k0+0..7 and k0+16..23 ; lane>=16: K = k0+8..15 and k0+24..31.
static __device__ __forceinline__ v16bf load_frag_a(const __bf16* rowPtr, int k0, int khalf) {
    Frag32B f;
    f.lo = *reinterpret_cast<const uint4*>(rowPtr + k0 + 8 * khalf);
    f.hi = *reinterpret_cast<const uint4*>(rowPtr + k0 + 16 + 8 * khalf);
    return __builtin_bit_cast(v16bf, f);
}

// B-matrix fragment (32x16 bf16) from a "Bt" layout where each output column's
// K-run is contiguous. lane holds column (lane&15); lane<16: K=k0+0..7 and k0+16..23,
// lane>=16: K=k0+8..15 and k0+24..31.
static __device__ __forceinline__ v16bf load_frag_b(const __bf16* colPtr, int k0, int khalf) {
    Frag32B f;
    f.lo = *reinterpret_cast<const uint4*>(colPtr + k0 + 8 * khalf);
    f.hi = *reinterpret_cast<const uint4*>(colPtr + k0 + 16 + 8 * khalf);
    return __builtin_bit_cast(v16bf, f);
}

static __device__ __forceinline__ v8f wmma_bf16(v16bf a, v16bf b, v8f c) {
    return __builtin_amdgcn_wmma_f32_16x16x32_bf16(false, a, false, b, (short)0, c, false, false);
}

// ---------------- Tensor Data Mover: 16x128 bf16 tile -> padded LDS ----------------
// D# per CDNA5 ISA. data_size=2B, tile 128x16, row stride
// 128 elements; LDS pad 8 DWORDs (16 halfs) after every 64 DWORDs (one 128-half
// row) -> padded LDS row stride 144 halfs, matching the As[] layout below.
static __device__ __forceinline__ void tdm_load_tile_16x128(const __bf16* gsrc,
                                                            __bf16* lds_dst) {
    unsigned lds_base = (unsigned)(unsigned long long)
        (__attribute__((address_space(3))) __bf16*)lds_dst;
    unsigned long long ga = (unsigned long long)gsrc;
    u32x4 g0;
    g0.x = 1u;                                              // count = 1 (valid D#)
    g0.y = lds_base;                                        // lds_addr (bytes)
    g0.z = (unsigned)(ga & 0xFFFFFFFFu);                    // global_addr[31:0]
    g0.w = (unsigned)((ga >> 32) & 0x01FFFFFFu) | (2u << 30); // addr[56:32] | type=2
    i32x8 g1;
    g1[0] = (int)((1u << 16) | (1u << 20) | (5u << 22) | (7u << 25)); // data_size=2B, pad_en, int=64DW, amt=8DW
    g1[1] = (int)(128u << 16);                              // tensor_dim0 = 128
    g1[2] = (int)(16u << 16);                               // tensor_dim1 = 16
    g1[3] = (int)(128u << 16);                              // tile_dim0 = 128
    g1[4] = 16;                                             // tile_dim1 = 16 (tile_dim2 = 0)
    g1[5] = 128;                                            // tensor_dim0_stride = 128
    g1[6] = 0; g1[7] = 0;
    i32x4 gz = {0, 0, 0, 0};
#if defined(__clang_major__) && (__clang_major__ >= 23)
    i32x8 z8 = {0, 0, 0, 0, 0, 0, 0, 0};
    __builtin_amdgcn_tensor_load_to_lds(g0, g1, gz, gz, z8, 0);
#else
    __builtin_amdgcn_tensor_load_to_lds(g0, g1, gz, gz, 0);
#endif
}

// ---------------- small elementwise kernels ----------------

__global__ void __launch_bounds__(256) init_h_kernel(const float* __restrict__ x,
                                                     float* __restrict__ hbuf) {
    int i = blockIdx.x * 256 + threadIdx.x;
    if (i < B_ * N_ * D_) hbuf[i] = x[i];
}

__global__ void __launch_bounds__(256) prep_proj_kernel(const float* __restrict__ projL,
                                                        __bf16* __restrict__ projb) {
    int i = blockIdx.x * 256 + threadIdx.x;
    if (i >= MP_ * DH_) return;
    int m = i >> 7, d = i & 127;
    projb[i] = (m < M_) ? (__bf16)projL[m * DH_ + d] : (__bf16)0.0f;
}

// LN1 + QKV projections (D=3 -> INNER=384), dn pre-scaled into q/k, diag terms.
__global__ void __launch_bounds__(128) ln_qkv_kernel(
    const float* __restrict__ hbuf, const float* __restrict__ s1, const float* __restrict__ bb1,
    const float* __restrict__ Wq, const float* __restrict__ bq,
    const float* __restrict__ Wk, const float* __restrict__ bk,
    const float* __restrict__ Wv, const float* __restrict__ bv,
    __bf16* __restrict__ qb, __bf16* __restrict__ kb, __bf16* __restrict__ vb,
    float* __restrict__ diagq, float* __restrict__ diagk) {
    const int r = blockIdx.x;            // b*N + n
    const int b = r / N_, n = r % N_;
    const int t = threadIdx.x;
    __shared__ float sdq[3], sdk[3];
    if (t < 3) { sdq[t] = 0.f; sdk[t] = 0.f; }
    float x0 = hbuf[r * 3 + 0], x1 = hbuf[r * 3 + 1], x2 = hbuf[r * 3 + 2];
    float mu = (x0 + x1 + x2) * (1.f / 3.f);
    float d0 = x0 - mu, d1 = x1 - mu, d2 = x2 - mu;
    float rs = rsqrtf((d0 * d0 + d1 * d1 + d2 * d2) * (1.f / 3.f) + 1e-5f);
    float z0 = d0 * rs * s1[0] + bb1[0];
    float z1 = d1 * rs * s1[1] + bb1[1];
    float z2 = d2 * rs * s1[2] + bb1[2];
    __syncthreads();
#pragma unroll
    for (int e = 0; e < 3; e++) {        // e == head index, t == dh
        int j = e * 128 + t;
        float q = z0 * Wq[j] + z1 * Wq[INNER_ + j] + z2 * Wq[2 * INNER_ + j] + bq[j];
        float k = z0 * Wk[j] + z1 * Wk[INNER_ + j] + z2 * Wk[2 * INNER_ + j] + bk[j];
        float v = z0 * Wv[j] + z1 * Wv[INNER_ + j] + z2 * Wv[2 * INNER_ + j] + bv[j];
        float qs = q * DN_, ks = k * DN_;
        size_t base = ((size_t)((b * H_ + e) * N_ + n)) * DH_ + t;
        qb[base] = (__bf16)qs; kb[base] = (__bf16)ks; vb[base] = (__bf16)v;
        atomicAdd(&sdq[e], qs * qs);
        atomicAdd(&sdk[e], ks * ks);
    }
    __syncthreads();
    if (t < 3) {
        diagq[(b * H_ + t) * N_ + n] = 0.5f * sdq[t];
        diagk[(b * H_ + t) * N_ + n] = 0.5f * sdk[t];
    }
}

// ---------------- WMMA feature kernels ----------------
// One workgroup: one (bh), 16 rows of n, all 640 feature columns.
// 8 waves x 5 col-tiles of 16. A tile staged to LDS by the TDM; B = projb
// direct global (proj rows are the columns of proj^T, contiguous).

__global__ void __launch_bounds__(256) k_rowmax_kernel(
    const __bf16* __restrict__ kb, const __bf16* __restrict__ projb,
    float* __restrict__ rowmaxk) {
    const int bh = blockIdx.y, n0 = blockIdx.x * 16;
    const int tid = threadIdx.x, lane = tid & 31, wave = tid >> 5;
    const int l15 = lane & 15, khalf = lane >> 4;
    __shared__ __align__(16) __bf16 As[16 * 144];
    __shared__ float redm[8][16];
    if (wave == 0) {
        tdm_load_tile_16x128(kb + ((size_t)(bh * N_ + n0)) * DH_, As);
        __builtin_amdgcn_s_wait_tensorcnt((short)0);
    }
    __syncthreads();
    float pmax[8];
#pragma unroll
    for (int i = 0; i < 8; i++) pmax[i] = -3.0e38f;
    const __bf16* arow = &As[l15 * 144];
    for (int t = 0; t < 5; t++) {
        int m0 = (wave * 5 + t) * 16;
        const __bf16* bcol = projb + (size_t)(m0 + l15) * DH_;
        v8f acc = zero8();
        v16bf af = load_frag_a(arow, 0, khalf);
        v16bf bf = load_frag_b(bcol, 0, khalf);
#pragma unroll
        for (int ks = 0; ks < 4; ks++) {
            v16bf ac = af, bc = bf;
            if (ks < 3) {                           // software pipeline next frags
                af = load_frag_a(arow, (ks + 1) * 32, khalf);
                bf = load_frag_b(bcol, (ks + 1) * 32, khalf);
            }
            acc = wmma_bf16(ac, bc, acc);
        }
#pragma unroll
        for (int i = 0; i < 8; i++) pmax[i] = fmaxf(pmax[i], acc[i]);
    }
#pragma unroll
    for (int i = 0; i < 8; i++)
        for (int m = 1; m < 16; m <<= 1) pmax[i] = fmaxf(pmax[i], __shfl_xor(pmax[i], m, 32));
    if (l15 == 0)
#pragma unroll
        for (int i = 0; i < 8; i++) redm[wave][i + 8 * khalf] = pmax[i];
    __syncthreads();
    if (tid < 16) {
        float mx = redm[0][tid];
#pragma unroll
        for (int w = 1; w < 8; w++) mx = fmaxf(mx, redm[w][tid]);
        rowmaxk[bh * N_ + n0 + tid] = mx;
    }
}

__global__ void __launch_bounds__(256) kmax_kernel(const float* __restrict__ rowmaxk,
                                                   float* __restrict__ kmax,
                                                   float* __restrict__ ksum) {
    const int bh = blockIdx.x, tid = threadIdx.x;
    __shared__ float s[256];
    float mx = -3.0e38f;
    for (int n = tid; n < N_; n += 256) mx = fmaxf(mx, rowmaxk[bh * N_ + n]);
    s[tid] = mx; __syncthreads();
    for (int off = 128; off > 0; off >>= 1) {
        if (tid < off) s[tid] = fmaxf(s[tid], s[tid + off]);
        __syncthreads();
    }
    if (tid == 0) kmax[bh] = s[0];
    for (int m = tid; m < MP_; m += 256) ksum[bh * MP_ + m] = 0.f;   // zero k_sum
}

__global__ void __launch_bounds__(256) k_feat_kernel(
    const __bf16* __restrict__ kb, const __bf16* __restrict__ projb,
    const float* __restrict__ diagk, const float* __restrict__ kmax,
    __bf16* __restrict__ kp, float* __restrict__ ksum) {
    const int bh = blockIdx.y, n0 = blockIdx.x * 16;
    const int tid = threadIdx.x, lane = tid & 31, wave = tid >> 5;
    const int l15 = lane & 15, khalf = lane >> 4;
    __shared__ __align__(16) __bf16 As[16 * 144];
    __shared__ float dks[16];
    if (wave == 0) {
        tdm_load_tile_16x128(kb + ((size_t)(bh * N_ + n0)) * DH_, As);
        __builtin_amdgcn_s_wait_tensorcnt((short)0);
    }
    if (tid >= 32 && tid < 48) dks[tid - 32] = diagk[bh * N_ + n0 + tid - 32];
    const float kmx = kmax[bh];
    __syncthreads();
    const __bf16* arow = &As[l15 * 144];
    for (int t = 0; t < 5; t++) {
        int m0 = (wave * 5 + t) * 16;
        const __bf16* bcol = projb + (size_t)(m0 + l15) * DH_;
        v8f acc = zero8();
        v16bf af = load_frag_a(arow, 0, khalf);
        v16bf bf = load_frag_b(bcol, 0, khalf);
#pragma unroll
        for (int ks = 0; ks < 4; ks++) {
            v16bf ac = af, bc = bf;
            if (ks < 3) {
                af = load_frag_a(arow, (ks + 1) * 32, khalf);
                bf = load_frag_b(bcol, (ks + 1) * 32, khalf);
            }
            acc = wmma_bf16(ac, bc, acc);
        }
        float csum = 0.f;
#pragma unroll
        for (int i = 0; i < 8; i++) {
            int row = i + 8 * khalf;
            float pv = KM_ * (__expf(acc[i] - dks[row] - kmx) + EPS_);
            kp[((size_t)(bh * N_ + n0 + row)) * MP_ + m0 + l15] = (__bf16)pv;
            csum += pv;
        }
        csum += __shfl_xor(csum, 16, 32);                 // lanes l and l+16 share column
        if (khalf == 0) atomicAdd(&ksum[bh * MP_ + m0 + l15], csum);
    }
}

__global__ void __launch_bounds__(256) q_feat_kernel(
    const __bf16* __restrict__ qb, const __bf16* __restrict__ projb,
    const float* __restrict__ diagq, const float* __restrict__ ksum,
    __bf16* __restrict__ qp, float* __restrict__ dinv) {
    const int bh = blockIdx.y, n0 = blockIdx.x * 16;
    const int tid = threadIdx.x, lane = tid & 31, wave = tid >> 5;
    const int l15 = lane & 15, khalf = lane >> 4;
    __shared__ __align__(16) __bf16 As[16 * 144];
    __shared__ float redm[8][16];
    __shared__ float rms[16], dqs[16], rsum[16];
    if (wave == 0) {
        tdm_load_tile_16x128(qb + ((size_t)(bh * N_ + n0)) * DH_, As);
        __builtin_amdgcn_s_wait_tensorcnt((short)0);
    }
    if (tid >= 32 && tid < 48) {
        dqs[tid - 32] = diagq[bh * N_ + n0 + tid - 32];
        rsum[tid - 32] = 0.f;
    }
    __syncthreads();
    const __bf16* arow = &As[l15 * 144];
    v8f accT[5];
    for (int t = 0; t < 5; t++) {
        int m0 = (wave * 5 + t) * 16;
        const __bf16* bcol = projb + (size_t)(m0 + l15) * DH_;
        v8f acc = zero8();
        v16bf af = load_frag_a(arow, 0, khalf);
        v16bf bf = load_frag_b(bcol, 0, khalf);
#pragma unroll
        for (int ks = 0; ks < 4; ks++) {
            v16bf ac = af, bc = bf;
            if (ks < 3) {
                af = load_frag_a(arow, (ks + 1) * 32, khalf);
                bf = load_frag_b(bcol, (ks + 1) * 32, khalf);
            }
            acc = wmma_bf16(ac, bc, acc);
        }
        accT[t] = acc;
    }
    float pmax[8];
#pragma unroll
    for (int i = 0; i < 8; i++) pmax[i] = -3.0e38f;
    for (int t = 0; t < 5; t++)
#pragma unroll
        for (int i = 0; i < 8; i++) pmax[i] = fmaxf(pmax[i], accT[t][i]);
#pragma unroll
    for (int i = 0; i < 8; i++)
        for (int m = 1; m < 16; m <<= 1) pmax[i] = fmaxf(pmax[i], __shfl_xor(pmax[i], m, 32));
    if (l15 == 0)
#pragma unroll
        for (int i = 0; i < 8; i++) redm[wave][i + 8 * khalf] = pmax[i];
    __syncthreads();
    if (tid < 16) {
        float mx = redm[0][tid];
#pragma unroll
        for (int w = 1; w < 8; w++) mx = fmaxf(mx, redm[w][tid]);
        rms[tid] = mx;
    }
    __syncthreads();
    float dsum[8];
#pragma unroll
    for (int i = 0; i < 8; i++) dsum[i] = 0.f;
    for (int t = 0; t < 5; t++) {
        int m0 = (wave * 5 + t) * 16;
        float kv = ksum[bh * MP_ + m0 + l15];
#pragma unroll
        for (int i = 0; i < 8; i++) {
            int row = i + 8 * khalf;
            float pv = KM_ * (__expf(accT[t][i] - dqs[row] - rms[row]) + EPS_);
            qp[((size_t)(bh * N_ + n0 + row)) * MP_ + m0 + l15] = (__bf16)pv;
            dsum[i] += pv * kv;
        }
    }
#pragma unroll
    for (int i = 0; i < 8; i++)
        for (int m = 1; m < 16; m <<= 1) dsum[i] += __shfl_xor(dsum[i], m, 32);
    if (l15 == 0)
#pragma unroll
        for (int i = 0; i < 8; i++) atomicAdd(&rsum[i + 8 * khalf], dsum[i]);
    __syncthreads();
    if (tid < 16) dinv[bh * N_ + n0 + tid] = 1.0f / rsum[tid];
}

// ---------------- ctx^T = v^T @ kp  (DH x MP, K = N) ----------------
__global__ void __launch_bounds__(256) ctx_gemm_kernel(
    const __bf16* __restrict__ vb, const __bf16* __restrict__ kp,
    __bf16* __restrict__ ctxb) {
    const int bh = blockIdx.z, m0g = blockIdx.x * 128, d0g = blockIdx.y * 64;
    const int tid = threadIdx.x, lane = tid & 31, wave = tid >> 5;
    const int l15 = lane & 15, khalf = lane >> 4;
    const int wy = wave >> 2, wx = wave & 3;                 // wy: d rows, wx: m cols
    __shared__ __align__(16) __bf16 Av[64 * 80];             // A(d,k): v transposed, K-step 64
    __shared__ __align__(16) __bf16 Bt[128 * 80];            // Bt(m,k): kp transposed
    v8f acc[2][2];
#pragma unroll
    for (int i = 0; i < 2; i++)
#pragma unroll
        for (int j = 0; j < 2; j++) acc[i][j] = zero8();
    const int n_off = tid >> 3, d8 = (tid & 7) * 8, m16 = (tid & 7) * 16;
    for (int k0 = 0; k0 < N_; k0 += 64) {
#pragma unroll
        for (int half = 0; half < 2; half++) {
            const int nn = n_off + half * 32;
            {   // stage A: Av[d][n] = v[n][d]
                U8H u;
                u.u = *reinterpret_cast<const uint4*>(
                    vb + ((size_t)(bh * N_ + k0 + nn)) * DH_ + d0g + d8);
#pragma unroll
                for (int j = 0; j < 8; j++) Av[(d8 + j) * 80 + nn] = u.h[j];
            }
            {   // stage B transposed: Bt[m][n] = kp[n][m]
                const __bf16* src = kp + ((size_t)(bh * N_ + k0 + nn)) * MP_ + m0g + m16;
                U8H u0, u1;
                u0.u = *reinterpret_cast<const uint4*>(src);
                u1.u = *reinterpret_cast<const uint4*>(src + 8);
#pragma unroll
                for (int j = 0; j < 8; j++) {
                    Bt[(m16 + j) * 80 + nn] = u0.h[j];
                    Bt[(m16 + 8 + j) * 80 + nn] = u1.h[j];
                }
            }
        }
        if (k0 + 64 < N_)
            __builtin_prefetch(kp + ((size_t)(bh * N_ + k0 + 64 + n_off)) * MP_ + m0g + m16, 0, 0);
        __syncthreads();
#pragma unroll
        for (int ks = 0; ks < 2; ks++) {
            v16bf a0 = load_frag_a(&Av[(wy * 32 + l15) * 80], ks * 32, khalf);
            v16bf a1 = load_frag_a(&Av[(wy * 32 + 16 + l15) * 80], ks * 32, khalf);
            v16bf b0 = load_frag_b(&Bt[(wx * 32 + l15) * 80], ks * 32, khalf);
            v16bf b1 = load_frag_b(&Bt[(wx * 32 + 16 + l15) * 80], ks * 32, khalf);
            acc[0][0] = wmma_bf16(a0, b0, acc[0][0]);
            acc[0][1] = wmma_bf16(a0, b1, acc[0][1]);
            acc[1][0] = wmma_bf16(a1, b0, acc[1][0]);
            acc[1][1] = wmma_bf16(a1, b1, acc[1][1]);
        }
        __syncthreads();
    }
#pragma unroll
    for (int i = 0; i < 2; i++)
#pragma unroll
        for (int j = 0; j < 2; j++)
#pragma unroll
            for (int e = 0; e < 8; e++) {
                int dd = d0g + wy * 32 + i * 16 + e + 8 * khalf;
                int mm = m0g + wx * 32 + j * 16 + l15;
                ctxb[((size_t)(bh * DH_ + dd)) * MP_ + mm] = (__bf16)acc[i][j][e];
            }
}

// ---------------- o = (qp @ ctx) * dinv  (N x DH, K = MP) ----------------
__global__ void __launch_bounds__(256) o_gemm_kernel(
    const __bf16* __restrict__ qp, const __bf16* __restrict__ ctxb,
    const float* __restrict__ dinv, float* __restrict__ obuf) {
    const int bh = blockIdx.y, b = bh / H_, h = bh % H_;
    const int n0 = blockIdx.x * 64;
    const int tid = threadIdx.x, lane = tid & 31, wave = tid >> 5;
    const int l15 = lane & 15, khalf = lane >> 4;
    const int wy = wave & 3, wx = wave >> 2;                 // 4 row tiles x 2 col halves
    const __bf16* arow = qp + ((size_t)(bh * N_ + n0 + wy * 16 + l15)) * MP_;
    const __bf16* bcol[4];
#pragma unroll
    for (int j = 0; j < 4; j++)
        bcol[j] = ctxb + ((size_t)(bh * DH_ + wx * 64 + j * 16 + l15)) * MP_;
    v8f acc[4];
#pragma unroll
    for (int j = 0; j < 4; j++) acc[j] = zero8();
    // double-buffered fragments across K steps
    v16bf af = load_frag_a(arow, 0, khalf);
    v16bf bf[4];
#pragma unroll
    for (int j = 0; j < 4; j++) bf[j] = load_frag_b(bcol[j], 0, khalf);
    for (int k0 = 0; k0 < MP_; k0 += 32) {
        v16bf ac = af;
        v16bf bc0 = bf[0], bc1 = bf[1], bc2 = bf[2], bc3 = bf[3];
        if (k0 + 32 < MP_) {
            af = load_frag_a(arow, k0 + 32, khalf);
#pragma unroll
            for (int j = 0; j < 4; j++) bf[j] = load_frag_b(bcol[j], k0 + 32, khalf);
        }
        acc[0] = wmma_bf16(ac, bc0, acc[0]);
        acc[1] = wmma_bf16(ac, bc1, acc[1]);
        acc[2] = wmma_bf16(ac, bc2, acc[2]);
        acc[3] = wmma_bf16(ac, bc3, acc[3]);
    }
    float dv[8];
#pragma unroll
    for (int e = 0; e < 8; e++) dv[e] = dinv[bh * N_ + n0 + wy * 16 + e + 8 * khalf];
#pragma unroll
    for (int j = 0; j < 4; j++) {
        int dcol = wx * 64 + j * 16 + l15;
#pragma unroll
        for (int e = 0; e < 8; e++) {
            int n = n0 + wy * 16 + e + 8 * khalf;
            obuf[((size_t)(b * N_ + n)) * INNER_ + h * DH_ + dcol] = acc[j][e] * dv[e];
        }
    }
}

// ---------------- out-proj + residual + LN2 + FFN (D=3, FF=12) ----------------
__global__ void __launch_bounds__(256) oproj_ffn_kernel(
    const float* __restrict__ obuf, const float* __restrict__ Wo, const float* __restrict__ bo,
    float* __restrict__ hbuf, const float* __restrict__ s2, const float* __restrict__ bb2,
    const float* __restrict__ W1, const float* __restrict__ b1f,
    const float* __restrict__ W2, const float* __restrict__ b2f) {
    int r = blockIdx.x * 256 + threadIdx.x;
    if (r >= B_ * N_) return;
    const float* orow = obuf + (size_t)r * INNER_;
    float a0 = bo[0], a1 = bo[1], a2 = bo[2];
    for (int j = 0; j < INNER_; j++) {
        float ov = orow[j];
        a0 += ov * Wo[j * 3 + 0]; a1 += ov * Wo[j * 3 + 1]; a2 += ov * Wo[j * 3 + 2];
    }
    float h0 = hbuf[r * 3 + 0] + a0;
    float h1 = hbuf[r * 3 + 1] + a1;
    float h2 = hbuf[r * 3 + 2] + a2;
    float mu = (h0 + h1 + h2) * (1.f / 3.f);
    float d0 = h0 - mu, d1 = h1 - mu, d2 = h2 - mu;
    float rs = rsqrtf((d0 * d0 + d1 * d1 + d2 * d2) * (1.f / 3.f) + 1e-5f);
    float z0 = d0 * rs * s2[0] + bb2[0];
    float z1 = d1 * rs * s2[1] + bb2[1];
    float z2 = d2 * rs * s2[2] + bb2[2];
    float f0 = 0.f, f1 = 0.f, f2 = 0.f;
#pragma unroll
    for (int f = 0; f < FF_; f++) {
        float u = z0 * W1[f] + z1 * W1[FF_ + f] + z2 * W1[2 * FF_ + f] + b1f[f];
        float g = 0.5f * u * (1.f + tanhf(0.7978845608028654f * (u + 0.044715f * u * u * u)));
        f0 += g * W2[f * 3 + 0]; f1 += g * W2[f * 3 + 1]; f2 += g * W2[f * 3 + 2];
    }
    hbuf[r * 3 + 0] = h0 + f0 + b2f[0];
    hbuf[r * 3 + 1] = h1 + f1 + b2f[1];
    hbuf[r * 3 + 2] = h2 + f2 + b2f[2];
}

__global__ void __launch_bounds__(256) decoder_kernel(
    const float* __restrict__ hbuf, const float* __restrict__ Wd,
    const float* __restrict__ bd, float* __restrict__ out) {
    int r = blockIdx.x * 256 + threadIdx.x;
    if (r >= B_ * N_) return;
    float s = hbuf[r * 3 + 0] * Wd[0] + hbuf[r * 3 + 1] * Wd[1] + hbuf[r * 3 + 2] * Wd[2] + bd[0];
    out[r] = 1.f / (1.f + __expf(-s));
}

// ---------------- host driver ----------------
extern "C" void kernel_launch(void* const* d_in, const int* in_sizes, int n_in,
                              void* d_out, int out_size, void* d_ws, size_t ws_size,
                              hipStream_t stream) {
    const float* x     = (const float*)d_in[0];
    const float* ln1_s = (const float*)d_in[1];
    const float* ln1_b = (const float*)d_in[2];
    const float* Wq    = (const float*)d_in[3];
    const float* bq    = (const float*)d_in[4];
    const float* Wk    = (const float*)d_in[5];
    const float* bk    = (const float*)d_in[6];
    const float* Wv    = (const float*)d_in[7];
    const float* bv    = (const float*)d_in[8];
    const float* Wo    = (const float*)d_in[9];
    const float* bo    = (const float*)d_in[10];
    const float* proj  = (const float*)d_in[11];
    const float* ln2_s = (const float*)d_in[12];
    const float* ln2_b = (const float*)d_in[13];
    const float* W1    = (const float*)d_in[14];
    const float* b1    = (const float*)d_in[15];
    const float* W2    = (const float*)d_in[16];
    const float* b2    = (const float*)d_in[17];
    const float* Wdec  = (const float*)d_in[18];
    const float* bdec  = (const float*)d_in[19];
    float* out = (float*)d_out;
    (void)in_sizes; (void)n_in; (void)out_size; (void)ws_size;

    char* p = (char*)d_ws;
    auto carve = [&](size_t bytes) -> char* {
        char* r = p; p += (bytes + 255) & ~(size_t)255; return r;
    };
    float*  hbuf    = (float*)carve(sizeof(float) * B_ * N_ * D_);
    __bf16* qb      = (__bf16*)carve(2ull * BH_ * N_ * DH_);
    __bf16* kb      = (__bf16*)carve(2ull * BH_ * N_ * DH_);
    __bf16* vb      = (__bf16*)carve(2ull * BH_ * N_ * DH_);
    float*  diagq   = (float*)carve(sizeof(float) * BH_ * N_);
    float*  diagk   = (float*)carve(sizeof(float) * BH_ * N_);
    float*  rowmaxk = (float*)carve(sizeof(float) * BH_ * N_);
    float*  kmaxb   = (float*)carve(sizeof(float) * 64);
    float*  ksum    = (float*)carve(sizeof(float) * BH_ * MP_);
    float*  dinv    = (float*)carve(sizeof(float) * BH_ * N_);
    __bf16* projb   = (__bf16*)carve(2ull * MP_ * DH_);
    __bf16* qp      = (__bf16*)carve(2ull * BH_ * N_ * MP_);
    __bf16* kp      = (__bf16*)carve(2ull * BH_ * N_ * MP_);
    __bf16* ctxb    = (__bf16*)carve(2ull * BH_ * DH_ * MP_);
    float*  obuf    = (float*)carve(sizeof(float) * B_ * N_ * INNER_);

    init_h_kernel<<<(B_ * N_ * D_ + 255) / 256, 256, 0, stream>>>(x, hbuf);

    for (int l = 0; l < L_; l++) {
        prep_proj_kernel<<<(MP_ * DH_ + 255) / 256, 256, 0, stream>>>(
            proj + (size_t)l * M_ * DH_, projb);
        ln_qkv_kernel<<<B_ * N_, 128, 0, stream>>>(
            hbuf, ln1_s + l * D_, ln1_b + l * D_,
            Wq + (size_t)l * D_ * INNER_, bq + (size_t)l * INNER_,
            Wk + (size_t)l * D_ * INNER_, bk + (size_t)l * INNER_,
            Wv + (size_t)l * D_ * INNER_, bv + (size_t)l * INNER_,
            qb, kb, vb, diagq, diagk);
        k_rowmax_kernel<<<dim3(N_ / 16, BH_), 256, 0, stream>>>(kb, projb, rowmaxk);
        kmax_kernel<<<BH_, 256, 0, stream>>>(rowmaxk, kmaxb, ksum);
        k_feat_kernel<<<dim3(N_ / 16, BH_), 256, 0, stream>>>(kb, projb, diagk, kmaxb, kp, ksum);
        ctx_gemm_kernel<<<dim3(MP_ / 128, DH_ / 64, BH_), 256, 0, stream>>>(vb, kp, ctxb);
        q_feat_kernel<<<dim3(N_ / 16, BH_), 256, 0, stream>>>(qb, projb, diagq, ksum, qp, dinv);
        o_gemm_kernel<<<dim3(N_ / 64, BH_), 256, 0, stream>>>(qp, ctxb, dinv, obuf);
        oproj_ffn_kernel<<<(B_ * N_ + 255) / 256, 256, 0, stream>>>(
            obuf, Wo + (size_t)l * INNER_ * D_, bo + (size_t)l * D_,
            hbuf, ln2_s + l * D_, ln2_b + l * D_,
            W1 + (size_t)l * D_ * FF_, b1 + (size_t)l * FF_,
            W2 + (size_t)l * FF_ * D_, b2 + (size_t)l * D_);
    }
    decoder_kernel<<<(B_ * N_ + 255) / 256, 256, 0, stream>>>(hbuf, Wdec, bdec, out);
}
